// SAB_13675175871087
// MI455X (gfx1250) — matrix operand order repaired and checked by reading, not compiled, not run on hardware
//
#include <hip/hip_runtime.h>
#include <hip/hip_bf16.h>

// ---------------------------------------------------------------------------
// Self-attention block (B=4, C=256, H=W=64) for gfx1250.
// All matmuls via v_wmma_f32_16x16x32_bf16 (bf16 operands, fp32 accumulate).
// Pipeline: [qkv GEMM] -> [flash attention w/ async-LDS double buffering]
//           -> [implicit-GEMM 3x3 conv + BN + ReLU + residual]
// Workspace layout (bytes):
//   qT  @ 0x000000  : (B,4096,32)  bf16  (1 MB)   n-major, c contiguous
//   kT  @ 0x100000  : (B,4096,32)  bf16  (1 MB)
//   v   @ 0x200000  : (B,256,4096) bf16  (8 MB)   c-major, n contiguous
//   O   @ 0xA00000  : (B,4096,256) bf16  (8 MB)   n-major, c contiguous
// ---------------------------------------------------------------------------

typedef __attribute__((ext_vector_type(16))) __bf16 v16bf;
typedef __attribute__((ext_vector_type(8)))  float  v8f;

#define B_   4
#define C_   256
#define CQ_  32
#define N_   4096
#define SCALE_ 0.17677669529663687f   // 1/sqrt(32)

// ---- CDNA5 async global->LDS path (guarded; falls back to sync copies) ----
#if defined(__has_builtin)
#  if __has_builtin(__builtin_amdgcn_global_load_async_to_lds_b128)
#    define HAVE_ASYNC_LDS 1
#  endif
#endif
#ifndef HAVE_ASYNC_LDS
#  define HAVE_ASYNC_LDS 0
#endif

#if HAVE_ASYNC_LDS
// builtin signature (from clang diagnostic): int4-vector pointers,
// global (AS1) source, LDS (AS3) destination, imm offset, imm cpol.
typedef int v4i_ __attribute__((vector_size(16)));
typedef __attribute__((address_space(1))) v4i_ GI4;
typedef __attribute__((address_space(3))) v4i_ LI4;
#endif

__device__ __forceinline__ void async_copy_b128(const void* g, void* l) {
#if HAVE_ASYNC_LDS
    // generic LDS pointer: low 32 bits are the LDS offset (ISA 10.2 aperture rules)
    GI4* gp = (GI4*)(unsigned long long)g;
    LI4* lp = (LI4*)(unsigned int)(unsigned long long)l;
    __builtin_amdgcn_global_load_async_to_lds_b128(gp, lp, 0, 0);
#else
    *(uint4*)l = *(const uint4*)g;
#endif
}

__device__ __forceinline__ void wait_async() {
#if HAVE_ASYNC_LDS
#  if __has_builtin(__builtin_amdgcn_s_wait_asynccnt)
    __builtin_amdgcn_s_wait_asynccnt(0);
#  else
    asm volatile("s_wait_asynccnt 0" ::: "memory");
#  endif
#endif
}

__device__ __forceinline__ unsigned short f2bf(float f) {
    unsigned u = __float_as_uint(f);
    u += 0x7FFFu + ((u >> 16) & 1u);          // round-to-nearest-even
    return (unsigned short)(u >> 16);
}

union FragBF { v16bf v; uint4 q[2]; };

__device__ __forceinline__ v8f wmma_bf16(const FragBF& a, const FragBF& b, v8f c) {
    return __builtin_amdgcn_wmma_f32_16x16x32_bf16(false, a.v, false, b.v,
                                                   (short)0, c, false, false);
}

// ===========================================================================
// Kernel 1: fused QKV projection.  D(320 x 16384) = W(320x256) * X(256x16384)
// rows 0-31 = Wq, 32-63 = Wk, 64-319 = Wv.  Block tile 64(M) x 128(cols).
// ===========================================================================
__global__ __launch_bounds__(256) void qkv_proj_kernel(
    const float* __restrict__ x,
    const float* __restrict__ Wq, const float* __restrict__ bq,
    const float* __restrict__ Wk, const float* __restrict__ bk,
    const float* __restrict__ Wv, const float* __restrict__ bv,
    unsigned short* __restrict__ qT, unsigned short* __restrict__ kT,
    unsigned short* __restrict__ vbf)
{
    __shared__ unsigned short ldsW[64 * 40];   // [m][k], stride 40 (16B-aligned frags)
    __shared__ unsigned short ldsX[128 * 40];  // [n][k]

    const int tid  = threadIdx.x;
    const int wave = tid >> 5, lane = tid & 31;
    const int hi   = lane >> 4, l15 = lane & 15;
    const int j0   = blockIdx.x * 128;
    const int b    = j0 >> 12, n0 = j0 & (N_ - 1);
    const int m0   = blockIdx.y * 64;
    const int wm   = (wave >> 1) * 16;
    const int wn   = (wave & 1) * 64;

    v8f acc[4];
    for (int t = 0; t < 4; ++t)
        for (int r = 0; r < 8; ++r) acc[t][r] = 0.f;

    for (int k0 = 0; k0 < C_; k0 += 32) {
        __syncthreads();
        // X tile -> LDS [n][k]  (coalesced along n)
        for (int idx = tid; idx < 128 * 32; idx += 256) {
            int kl = idx >> 7, nl = idx & 127;
            ldsX[nl * 40 + kl] = f2bf(x[(size_t)(b * C_ + k0 + kl) * N_ + n0 + nl]);
        }
        // prefetch next X k-slice (one 256B-line hint per row-half)
        if (k0 + 32 < C_ && tid < 64) {
            int kl = tid >> 1, half = tid & 1;
            __builtin_prefetch(&x[(size_t)(b * C_ + k0 + 32 + kl) * N_ + n0 + half * 64], 0, 1);
        }
        // W tile -> LDS [m][k]  (coalesced along c)
        for (int idx = tid; idx < 64 * 32; idx += 256) {
            int ml = idx >> 5, kl = idx & 31;
            int mg = m0 + ml;
            const float* src; int row;
            if (mg < 32)      { src = Wq; row = mg; }
            else if (mg < 64) { src = Wk; row = mg - 32; }
            else              { src = Wv; row = mg - 64; }
            ldsW[ml * 40 + kl] = f2bf(src[row * C_ + k0 + kl]);
        }
        __syncthreads();

        FragBF a;  // A: 16x32, lanes 0-15 K {0-7,16-23}, lanes 16-31 K {8-15,24-31}
        {
            const unsigned short* wb = &ldsW[(wm + l15) * 40];
            a.q[0] = *(const uint4*)(wb + (hi ? 8 : 0));
            a.q[1] = *(const uint4*)(wb + (hi ? 24 : 16));
        }
        for (int t = 0; t < 4; ++t) {
            const unsigned short* bb = &ldsX[(wn + t * 16 + l15) * 40 + hi * 8];
            FragBF bf;
            bf.q[0] = *(const uint4*)(bb);
            bf.q[1] = *(const uint4*)(bb + 16);
            acc[t] = wmma_bf16(a, bf, acc[t]);
        }
    }

    // epilogue: bias + scatter to qT / kT / v
    for (int t = 0; t < 4; ++t) {
        int n = n0 + wn + t * 16 + l15;
        for (int r = 0; r < 8; ++r) {
            int mg = m0 + wm + r + hi * 8;
            float val = acc[t][r];
            if (mg < 32) {
                val += bq[mg];
                qT[(size_t)(b * N_ + n) * CQ_ + mg] = f2bf(val);
            } else if (mg < 64) {
                val += bk[mg - 32];
                kT[(size_t)(b * N_ + n) * CQ_ + (mg - 32)] = f2bf(val);
            } else {
                val += bv[mg - 64];
                vbf[(size_t)(b * C_ + (mg - 64)) * N_ + n] = f2bf(val);
            }
        }
    }
}

// ===========================================================================
// Kernel 2: flash attention, 64 queries x all 4096 keys per workgroup.
// K/V tiles double-buffered in LDS via async global->LDS copies, overlapped
// with S-GEMM + softmax + PV-GEMM of the current tile.
// ===========================================================================
#define KSTR 40   // sK row stride (ushorts)
#define VSTR 72   // sV row stride (ushorts)

__global__ __launch_bounds__(256) void attn_kernel(
    const unsigned short* __restrict__ qT, const unsigned short* __restrict__ kT,
    const unsigned short* __restrict__ vbf, unsigned short* __restrict__ obf)
{
    __shared__ unsigned short sK[2][64 * KSTR];    // K tile [m][c]
    __shared__ unsigned short sV[2][256 * VSTR];   // V tile [c][m]
    __shared__ float          sS[64 * 65];         // scores (stride 65)
    __shared__ unsigned short sP[64 * 72];         // bf16 probabilities
    __shared__ float sRed[256];
    __shared__ float sM[64], sL[64], sAlpha[64];

    const int tid  = threadIdx.x;
    const int wave = tid >> 5, lane = tid & 31;
    const int hi   = lane >> 4, l15 = lane & 15;
    const int n0   = blockIdx.x * 64;
    const int b    = blockIdx.y;
    const int ti   = wave >> 1;             // n-tile (0..3), shared by S and PV
    const int chHalf = (wave & 1) * 128;    // PV channel half
    const int row  = tid >> 2, quad = tid & 3;   // softmax mapping

    v8f acc[8];
    for (int t = 0; t < 8; ++t)
        for (int r = 0; r < 8; ++r) acc[t][r] = 0.f;

    if (tid < 64) { sM[tid] = -3.0e38f; sL[tid] = 0.f; }

    // persistent Q^T fragment for this wave's 16 query rows
    FragBF aQ;
    {
        const unsigned short* base = qT + (size_t)(b * N_ + n0 + ti * 16 + l15) * CQ_;
        aQ.q[0] = *(const uint4*)(base + (hi ? 8 : 0));
        aQ.q[1] = *(const uint4*)(base + (hi ? 24 : 16));
    }

    // ---- stage helpers (16B per async op) ----
    auto stage_k = [&](int m0k, unsigned short* dst) {
        int m = tid >> 2, ch = tid & 3;                       // 64 rows x 4 chunks
        const unsigned short* g = kT + (size_t)(b * N_ + m0k + m) * CQ_ + ch * 8;
        async_copy_b128(g, dst + m * KSTR + ch * 8);
    };
    auto stage_v = [&](int m0k, unsigned short* dst) {
        for (int it = 0; it < 8; ++it) {                      // 256 rows x 8 chunks
            int idx = it * 256 + tid;
            int c = idx >> 3, ch = idx & 7;
            const unsigned short* g = vbf + (size_t)(b * C_ + c) * N_ + m0k + ch * 8;
            async_copy_b128(g, dst + c * VSTR + ch * 8);
        }
    };

    // prologue: stage tile 0
    stage_k(0, sK[0]);
    stage_v(0, sV[0]);
    wait_async();
    __syncthreads();

    int cur = 0;
    for (int m0 = 0; m0 < N_; m0 += 64) {
        // ---- kick off async staging of the NEXT tile (overlapped) ----
        if (m0 + 64 < N_) {
            stage_k(m0 + 64, sK[cur ^ 1]);
            stage_v(m0 + 64, sV[cur ^ 1]);
        }

        // ---- S tiles (2 per wave) from LDS K, scaled, into sS ----
        for (int jj = 0; jj < 2; ++jj) {
            int tj = 2 * (wave & 1) + jj;
            FragBF bK;
            const unsigned short* kb = &sK[cur][(tj * 16 + l15) * KSTR + hi * 8];
            bK.q[0] = *(const uint4*)(kb);
            bK.q[1] = *(const uint4*)(kb + 16);
            v8f s;
            for (int r = 0; r < 8; ++r) s[r] = 0.f;
            s = wmma_bf16(aQ, bK, s);
            int col = tj * 16 + l15;
            for (int r = 0; r < 8; ++r)
                sS[(ti * 16 + r + hi * 8) * 65 + col] = s[r] * SCALE_;
        }
        __syncthreads();

        // ---- online softmax, 4 threads per row ----
        const float* rowp = &sS[row * 65 + quad * 16];
        float pm = -3.0e38f;
        for (int i = 0; i < 16; ++i) pm = fmaxf(pm, rowp[i]);
        sRed[tid] = pm;
        __syncthreads();
        if (tid < 64) {
            float mOld = sM[tid];
            float mNew = fmaxf(fmaxf(fmaxf(sRed[tid * 4 + 0], sRed[tid * 4 + 1]),
                                     fmaxf(sRed[tid * 4 + 2], sRed[tid * 4 + 3])), mOld);
            sAlpha[tid] = __expf(mOld - mNew);
            sM[tid] = mNew;
        }
        __syncthreads();
        float mNew = sM[row];
        float ps = 0.f;
        unsigned short* prow = &sP[row * 72 + quad * 16];
        for (int i = 0; i < 16; ++i) {
            float p = __expf(rowp[i] - mNew);
            ps += p;
            prow[i] = f2bf(p);
        }
        sRed[tid] = ps;
        __syncthreads();
        if (tid < 64) {
            sL[tid] = sL[tid] * sAlpha[tid]
                    + (sRed[tid * 4 + 0] + sRed[tid * 4 + 1])
                    + (sRed[tid * 4 + 2] + sRed[tid * 4 + 3]);
        }
        __syncthreads();

        // ---- rescale running O, then O += P * V^T (V from LDS) ----
        float al[8];
        for (int r = 0; r < 8; ++r) al[r] = sAlpha[ti * 16 + r + hi * 8];
        for (int t = 0; t < 8; ++t)
            for (int r = 0; r < 8; ++r) acc[t][r] *= al[r];

        for (int kk = 0; kk < 2; ++kk) {       // m-chunks of 32
            FragBF aP;
            const unsigned short* pb = &sP[(ti * 16 + l15) * 72 + kk * 32];
            aP.q[0] = *(const uint4*)(pb + (hi ? 8 : 0));
            aP.q[1] = *(const uint4*)(pb + (hi ? 24 : 16));
            for (int t = 0; t < 8; ++t) {
                int c = chHalf + t * 16 + l15;
                const unsigned short* vb = &sV[cur][c * VSTR + kk * 32 + hi * 8];
                FragBF bV;
                bV.q[0] = *(const uint4*)(vb);
                bV.q[1] = *(const uint4*)(vb + 16);
                acc[t] = wmma_bf16(aP, bV, acc[t]);
            }
        }

        // drain next-tile async copies, make them visible to all waves
        wait_async();
        __syncthreads();
        cur ^= 1;
    }

    // ---- normalize by l and store O^T (n-major, c contiguous) ----
    float linv[8];
    for (int r = 0; r < 8; ++r) linv[r] = 1.0f / sL[ti * 16 + r + hi * 8];
    for (int t = 0; t < 8; ++t) {
        int c = chHalf + t * 16 + l15;
        for (int r = 0; r < 8; ++r) {
            int n = n0 + ti * 16 + r + hi * 8;
            obf[(size_t)(b * N_ + n) * C_ + c] = f2bf(acc[t][r] * linv[r]);
        }
    }
}

// ===========================================================================
// Kernel 3: 3x3 conv (implicit GEMM over 9 taps x C) + bias + BN + ReLU + res.
// ===========================================================================
__global__ __launch_bounds__(256) void conv_bn_relu_kernel(
    const unsigned short* __restrict__ obf, const float* __restrict__ Wo,
    const float* __restrict__ bo, const float* __restrict__ gamma,
    const float* __restrict__ beta, const float* __restrict__ mean,
    const float* __restrict__ var, const float* __restrict__ x,
    float* __restrict__ out)
{
    __shared__ unsigned short ldsW[64 * 40];

    const int tid  = threadIdx.x;
    const int wave = tid >> 5, lane = tid & 31;
    const int hi   = lane >> 4, l15 = lane & 15;
    const int p0   = blockIdx.x * 128;
    const int o0   = blockIdx.y * 64;
    const int b    = blockIdx.z;
    const int wm   = (wave >> 1) * 16;
    const int wn   = (wave & 1) * 64;

    v8f acc[4];
    for (int t = 0; t < 4; ++t)
        for (int r = 0; r < 8; ++r) acc[t][r] = 0.f;

    for (int dy = 0; dy < 3; ++dy)
    for (int dx = 0; dx < 3; ++dx) {
        for (int c0 = 0; c0 < C_; c0 += 32) {
            __syncthreads();
            for (int idx = tid; idx < 64 * 32; idx += 256) {
                int ol = idx >> 5, cl = idx & 31;
                ldsW[ol * 40 + cl] =
                    f2bf(Wo[((size_t)(o0 + ol) * C_ + c0 + cl) * 9 + dy * 3 + dx]);
            }
            __syncthreads();

            FragBF a;
            const unsigned short* wb = &ldsW[(wm + l15) * 40];
            a.q[0] = *(const uint4*)(wb + (hi ? 8 : 0));
            a.q[1] = *(const uint4*)(wb + (hi ? 24 : 16));

            for (int t = 0; t < 4; ++t) {
                int p  = p0 + wn + t * 16 + l15;
                int yy = p >> 6, xx = p & 63;
                int py = yy + dy - 1, px = xx + dx - 1;
                FragBF bf;
                if (py >= 0 && py < 64 && px >= 0 && px < 64) {
                    const unsigned short* ob =
                        obf + (size_t)(b * N_ + py * 64 + px) * C_ + c0 + hi * 8;
                    bf.q[0] = *(const uint4*)(ob);
                    bf.q[1] = *(const uint4*)(ob + 16);
                } else {
                    uint4 z; z.x = z.y = z.z = z.w = 0u;
                    bf.q[0] = z; bf.q[1] = z;
                }
                acc[t] = wmma_bf16(a, bf, acc[t]);
            }
        }
    }

    // fused bias + BN + ReLU + residual, fp32 out
    for (int r = 0; r < 8; ++r) {
        int o = o0 + wm + r + hi * 8;
        float inv   = gamma[o] * rsqrtf(var[o] + 1e-5f);
        float shift = beta[o] - mean[o] * inv;
        float bias  = bo[o];
        for (int t = 0; t < 4; ++t) {
            int p = p0 + wn + t * 16 + l15;
            size_t idx = (size_t)(b * C_ + o) * N_ + p;
            float val = (acc[t][r] + bias) * inv + shift;
            val = fmaxf(val, 0.f);
            out[idx] = val + x[idx];
        }
    }
}

// ===========================================================================
extern "C" void kernel_launch(void* const* d_in, const int* in_sizes, int n_in,
                              void* d_out, int out_size, void* d_ws, size_t ws_size,
                              hipStream_t stream) {
    const float* x     = (const float*)d_in[0];
    const float* Wq    = (const float*)d_in[1];
    const float* bq    = (const float*)d_in[2];
    const float* Wk    = (const float*)d_in[3];
    const float* bk    = (const float*)d_in[4];
    const float* Wv    = (const float*)d_in[5];
    const float* bv    = (const float*)d_in[6];
    const float* Wo    = (const float*)d_in[7];
    const float* bo    = (const float*)d_in[8];
    const float* gamma = (const float*)d_in[9];
    const float* beta  = (const float*)d_in[10];
    const float* mean  = (const float*)d_in[11];
    const float* var   = (const float*)d_in[12];
    float* out = (float*)d_out;

    char* wsb = (char*)d_ws;
    unsigned short* qT  = (unsigned short*)(wsb);
    unsigned short* kT  = (unsigned short*)(wsb + (1u << 20));
    unsigned short* vbf = (unsigned short*)(wsb + (2u << 20));
    unsigned short* obf = (unsigned short*)(wsb + (10u << 20));

    dim3 g1(128, 5, 1);                 // 16384/128 col tiles x 320/64 row tiles
    qkv_proj_kernel<<<g1, 256, 0, stream>>>(x, Wq, bq, Wk, bk, Wv, bv, qT, kT, vbf);

    dim3 g2(64, 4, 1);                  // 4096/64 query tiles x B
    attn_kernel<<<g2, 256, 0, stream>>>(qT, kT, vbf, obf);

    dim3 g3(32, 4, 4);                  // 4096/128 pixel tiles x 256/64 o tiles x B
    conv_bn_relu_kernel<<<g3, 256, 0, stream>>>(obf, Wo, bo, gamma, beta, mean,
                                                var, x, out);
}
